// CLIPLoss_22333829939494
// MI455X (gfx1250) — matrix operand order repaired and checked by reading, not compiled, not run on hardware
//
#include <hip/hip_runtime.h>
#include <math.h>

typedef __bf16 bf16;
typedef __attribute__((ext_vector_type(16))) __bf16 v16bf;
typedef __attribute__((ext_vector_type(8)))  __bf16 v8bf;
typedef __attribute__((ext_vector_type(8)))  float  v8f;

#define BN 16384
#define DN 256
#define ND (BN * DN)

union V16 { v16bf v; v8bf h[2]; };

// ---- packed (max,sum) log-sum-exp atomic merge: lo32=max bits, hi32=sum bits
__device__ __forceinline__ void atomicLseMerge(unsigned long long* p, float m1, float s1) {
    unsigned long long old = *(volatile unsigned long long*)p;
    unsigned long long assumed;
    do {
        assumed = old;
        float m0 = __uint_as_float((unsigned int)(assumed & 0xffffffffull));
        float s0 = __uint_as_float((unsigned int)(assumed >> 32));
        float m  = fmaxf(m0, m1);
        float s  = s0 * __expf(m0 - m) + s1 * __expf(m1 - m);  // m0=-inf,s0=0 -> s1
        unsigned long long np =
            ((unsigned long long)__float_as_uint(s) << 32) | (unsigned long long)__float_as_uint(m);
        old = atomicCAS(p, assumed, np);
    } while (old != assumed);
}

// ---- prep: bf16 hi/lo split of both inputs + stat init + scale ---------------
__global__ __launch_bounds__(256) void k_prep(const float* __restrict__ zs,
                                              const float* __restrict__ zr,
                                              const float* __restrict__ ls,
                                              bf16* __restrict__ AHI, bf16* __restrict__ ALO,
                                              bf16* __restrict__ BHI, bf16* __restrict__ BLO,
                                              unsigned long long* __restrict__ rstat,
                                              unsigned long long* __restrict__ cstat,
                                              float* __restrict__ scalep) {
    int idx = blockIdx.x * 256 + threadIdx.x;
    float xa = zs[idx];
    bf16 ah = (bf16)xa;
    AHI[idx] = ah;
    ALO[idx] = (bf16)(xa - (float)ah);
    float xb = zr[idx];
    bf16 bh = (bf16)xb;
    BHI[idx] = bh;
    BLO[idx] = (bf16)(xb - (float)bh);
    if (idx < BN) {
        const unsigned long long init = 0x00000000FF800000ull;  // max=-inf, sum=0
        rstat[idx] = init;
        cstat[idx] = init;
    }
    if (idx == 0) scalep[0] = fminf(__expf(ls[0]), 100.0f);
}

// ---- single-pass GEMM (bf16x3) + online row/col LSE stats -------------------
// Each wave: 32(rows) x 64(cols) strip = 2x4 WMMA tiles; K-chunks of 32 with
// 3 bf16 WMMAs per tile (hi/lo split).  B fragments reused across both row
// groups in-register; 8 waves/block share the same 64-col B strip via cache.
__global__ __launch_bounds__(256) void k_gemm(const bf16* __restrict__ AHI, const bf16* __restrict__ ALO,
                                              const bf16* __restrict__ BHI, const bf16* __restrict__ BLO,
                                              unsigned long long* __restrict__ rstat,
                                              unsigned long long* __restrict__ cstat,
                                              const float* __restrict__ scalep) {
    const int lane = threadIdx.x & 31;
    const int wave = threadIdx.x >> 5;
    const int tj   = blockIdx.x >> 6;                    // 0..255 : 64-col strip
    const int ti   = ((blockIdx.x & 63) << 3) + wave;    // 0..511 : 32-row tile
    const int i0   = ti << 5;
    const int j0   = tj << 6;
    const int r    = lane & 15;
    const int hi16 = lane >> 4;
    const int kha  = hi16 << 3;   // A frag K sub-offset: 0 or 8
    const int khb  = hi16 << 3;   // B frag K sub-offset: 0 or 8
    const int half8 = hi16 << 3;  // C layout: row offset of this lane half

    const bf16* arh0 = AHI + (size_t)(i0 + r) * DN;
    const bf16* arl0 = ALO + (size_t)(i0 + r) * DN;
    const bf16* arh1 = arh0 + 16 * DN;
    const bf16* arl1 = arl0 + 16 * DN;
    const bf16* brh  = BHI + (size_t)(j0 + r) * DN;
    const bf16* brl  = BLO + (size_t)(j0 + r) * DN;

    v8f acc[2][4];
#pragma unroll
    for (int g = 0; g < 2; ++g)
#pragma unroll
        for (int t = 0; t < 4; ++t) acc[g][t] = v8f{};

#pragma unroll 2
    for (int k0 = 0; k0 < DN; k0 += 32) {
        V16 ah0, al0, ah1, al1;
        ah0.h[0] = *(const v8bf*)(arh0 + k0 + kha);
        ah0.h[1] = *(const v8bf*)(arh0 + k0 + 16 + kha);
        al0.h[0] = *(const v8bf*)(arl0 + k0 + kha);
        al0.h[1] = *(const v8bf*)(arl0 + k0 + 16 + kha);
        ah1.h[0] = *(const v8bf*)(arh1 + k0 + kha);
        ah1.h[1] = *(const v8bf*)(arh1 + k0 + 16 + kha);
        al1.h[0] = *(const v8bf*)(arl1 + k0 + kha);
        al1.h[1] = *(const v8bf*)(arl1 + k0 + 16 + kha);
#pragma unroll
        for (int t = 0; t < 4; ++t) {
            const bf16* ph = brh + t * (16 * DN) + k0 + khb;
            const bf16* pl = brl + t * (16 * DN) + k0 + khb;
            V16 bh, bl;
            bh.h[0] = *(const v8bf*)(ph);
            bh.h[1] = *(const v8bf*)(ph + 16);
            bl.h[0] = *(const v8bf*)(pl);
            bl.h[1] = *(const v8bf*)(pl + 16);
            // a*b ~= lo_a*hi_b + hi_a*lo_b + hi_a*hi_b  (f32 accumulate)
            acc[0][t] = __builtin_amdgcn_wmma_f32_16x16x32_bf16(false, al0.v, false, bh.v, (short)0, acc[0][t], false, false);
            acc[0][t] = __builtin_amdgcn_wmma_f32_16x16x32_bf16(false, ah0.v, false, bl.v, (short)0, acc[0][t], false, false);
            acc[0][t] = __builtin_amdgcn_wmma_f32_16x16x32_bf16(false, ah0.v, false, bh.v, (short)0, acc[0][t], false, false);
            acc[1][t] = __builtin_amdgcn_wmma_f32_16x16x32_bf16(false, al1.v, false, bh.v, (short)0, acc[1][t], false, false);
            acc[1][t] = __builtin_amdgcn_wmma_f32_16x16x32_bf16(false, ah1.v, false, bl.v, (short)0, acc[1][t], false, false);
            acc[1][t] = __builtin_amdgcn_wmma_f32_16x16x32_bf16(false, ah1.v, false, bh.v, (short)0, acc[1][t], false, false);
        }
    }

    const float s = scalep[0];
#pragma unroll
    for (int g = 0; g < 2; ++g)
#pragma unroll
        for (int t = 0; t < 4; ++t)
#pragma unroll
            for (int rr = 0; rr < 8; ++rr) acc[g][t][rr] *= s;

    // ---- row-wise local (max, sum-exp) over the 64 columns ------------------
#pragma unroll
    for (int g = 0; g < 2; ++g) {
#pragma unroll
        for (int rr = 0; rr < 8; ++rr) {
            float m = fmaxf(fmaxf(acc[g][0][rr], acc[g][1][rr]),
                            fmaxf(acc[g][2][rr], acc[g][3][rr]));
#pragma unroll
            for (int off = 1; off < 16; off <<= 1)
                m = fmaxf(m, __shfl_xor(m, off, 32));
            float e = __expf(acc[g][0][rr] - m) + __expf(acc[g][1][rr] - m) +
                      __expf(acc[g][2][rr] - m) + __expf(acc[g][3][rr] - m);
#pragma unroll
            for (int off = 1; off < 16; off <<= 1)
                e += __shfl_xor(e, off, 32);
            if (r == 0) atomicLseMerge(rstat + i0 + 16 * g + rr + half8, m, e);
        }
    }
    // ---- col-wise local (max, sum-exp) over the 32 rows ---------------------
#pragma unroll
    for (int t = 0; t < 4; ++t) {
        float m = acc[0][t][0];
#pragma unroll
        for (int rr = 1; rr < 8; ++rr) m = fmaxf(m, acc[0][t][rr]);
#pragma unroll
        for (int rr = 0; rr < 8; ++rr) m = fmaxf(m, acc[1][t][rr]);
        m = fmaxf(m, __shfl_xor(m, 16, 32));
        float e = 0.0f;
#pragma unroll
        for (int rr = 0; rr < 8; ++rr)
            e += __expf(acc[0][t][rr] - m) + __expf(acc[1][t][rr] - m);
        e += __shfl_xor(e, 16, 32);
        if (lane < 16) atomicLseMerge(cstat + j0 + 16 * t + lane, m, e);
    }
}

// ---- diagonal in full f32: one wave per row ---------------------------------
__global__ __launch_bounds__(256) void k_diag(const float* __restrict__ A,
                                              const float* __restrict__ Bm,
                                              const float* __restrict__ scalep,
                                              float* __restrict__ diag) {
    const int lane = threadIdx.x & 31;
    const int row  = blockIdx.x * 8 + (threadIdx.x >> 5);
    const float* a = A  + (size_t)row * DN;
    const float* b = Bm + (size_t)row * DN;
    float sum = 0.0f;
#pragma unroll
    for (int d = lane; d < DN; d += 32) sum += a[d] * b[d];
#pragma unroll
    for (int off = 1; off < 32; off <<= 1)
        sum += __shfl_xor(sum, off, 32);
    if (lane == 0) diag[row] = scalep[0] * sum;
}

// ---- final reduction --------------------------------------------------------
__global__ __launch_bounds__(256) void k_final(const unsigned long long* __restrict__ rstat,
                                               const unsigned long long* __restrict__ cstat,
                                               const float* __restrict__ diag,
                                               float* __restrict__ out) {
    __shared__ float sm[256];
    float p = 0.0f;
    for (int i = threadIdx.x; i < BN; i += 256) {
        unsigned long long rv = rstat[i], cv = cstat[i];
        float rm = __uint_as_float((unsigned int)(rv & 0xffffffffull));
        float rs = __uint_as_float((unsigned int)(rv >> 32));
        float cm = __uint_as_float((unsigned int)(cv & 0xffffffffull));
        float cs = __uint_as_float((unsigned int)(cv >> 32));
        p += (rm + logf(rs)) + (cm + logf(cs)) - 2.0f * diag[i];
    }
    sm[threadIdx.x] = p;
    __syncthreads();
    for (int off = 128; off > 0; off >>= 1) {
        if (threadIdx.x < off) sm[threadIdx.x] += sm[threadIdx.x + off];
        __syncthreads();
    }
    if (threadIdx.x == 0) {
        float loss = 0.5f * sm[0] / (float)BN;
        out[0] = loss;
        out[1] = loss;
    }
}

// ---- launcher ---------------------------------------------------------------
// workspace: AHI|ALO|BHI|BLO (bf16, 4*ND elems = 33.5 MB), then
// row_stat u64[BN], col_stat u64[BN], diag f32[BN], scale f32[1]  (~34 MB total)
extern "C" void kernel_launch(void* const* d_in, const int* in_sizes, int n_in,
                              void* d_out, int out_size, void* d_ws, size_t ws_size,
                              hipStream_t stream) {
    (void)in_sizes; (void)n_in; (void)out_size; (void)ws_size;
    const float* z_schema    = (const float*)d_in[0];
    const float* z_seal      = (const float*)d_in[1];
    const float* logit_scale = (const float*)d_in[2];
    float* out = (float*)d_out;

    char* w = (char*)d_ws;
    bf16* AHI = (bf16*)w;
    bf16* ALO = AHI + ND;
    bf16* BHI = ALO + ND;
    bf16* BLO = BHI + ND;
    unsigned long long* rstat = (unsigned long long*)(w + (size_t)4 * ND * sizeof(bf16));
    unsigned long long* cstat = rstat + BN;
    float* diag   = (float*)(cstat + BN);
    float* scalep = diag + BN;

    k_prep<<<ND / 256, 256, 0, stream>>>(z_schema, z_seal, logit_scale,
                                         AHI, ALO, BHI, BLO, rstat, cstat, scalep);

    // 512 row-tiles (32 rows) x 256 col-strips (64 cols) = 131072 wave-strips
    // / 8 waves per block = 16384 blocks
    k_gemm<<<16384, 256, 0, stream>>>(AHI, ALO, BHI, BLO, rstat, cstat, scalep);

    k_diag<<<BN / 8, 256, 0, stream>>>(z_schema, z_seal, scalep, diag);
    k_final<<<1, 256, 0, stream>>>(rstat, cstat, diag, out);
}
